// DeformConv_8186207666989
// MI455X (gfx1250) — compile-verified, matrix operand order checked by reading, not run on hardware
//
#include <hip/hip_runtime.h>
#include <hip/hip_bf16.h>
#include <math.h>

typedef __attribute__((ext_vector_type(16))) _Float16 v16h;
typedef __attribute__((ext_vector_type(8)))  _Float16 v8h;
typedef __attribute__((ext_vector_type(8)))  float    v8f;

#define KKPTS 9
#define C_IN  256
#define O_OUT 256
#define HDIM  64
#define WDIM  64
#define BDIM  4
#define KRED  (C_IN * KKPTS)          // 2304
#define MTOT  (BDIM * HDIM * WDIM)    // 16384
#define KSLAB 256                     // K-chunk staged in LDS (weights)
#define NSLAB (KRED / KSLAB)          // 9
#define OTIL  4                       // o-tiles per wave (64 o rows / block)

// ---------------------------------------------------------------------------
// Kernel 0: weight cast to f16 (layout [O][C*9] == w.reshape(O, C*9)) and
// BN fold: scale = gamma/sqrt(var+eps); shift = b*scale + beta - mean*scale
// ---------------------------------------------------------------------------
__global__ void prep_kernel(const float* __restrict__ w, const float* __restrict__ b,
                            const float* __restrict__ gamma, const float* __restrict__ beta,
                            const float* __restrict__ rmean, const float* __restrict__ rvar,
                            _Float16* __restrict__ BT, float* __restrict__ scale,
                            float* __restrict__ shift)
{
    int tid = blockIdx.x * 256 + threadIdx.x;
    if (tid < O_OUT * KRED) BT[tid] = (_Float16)w[tid];
    if (tid < O_OUT) {
        float inv = gamma[tid] * rsqrtf(rvar[tid] + 1e-5f);
        scale[tid] = inv;
        shift[tid] = b[tid] * inv + beta[tid] - rmean[tid] * inv;
    }
}

// ---------------------------------------------------------------------------
// Kernel 1: offset conv  om[b][j][h][w] = b_off[j] + sum_{c,dy,dx} x*w_off
// j is block-uniform -> weight loads become scalar loads.
// ---------------------------------------------------------------------------
__global__ void offconv_kernel(const float* __restrict__ x, const float* __restrict__ w_off,
                               const float* __restrict__ b_off, float* __restrict__ om)
{
    int hw = blockIdx.x * 256 + threadIdx.x;   // gridDim.x = 16 -> 4096
    int j  = blockIdx.y;                       // 0..26
    int b  = blockIdx.z;                       // 0..3
    int h  = hw >> 6, wc = hw & 63;

    float acc = b_off[j];
    const float* xb = x + ((size_t)b * C_IN << 12);
    const float* wj = w_off + (size_t)j * C_IN * 9;
    for (int c = 0; c < C_IN; ++c) {
        const float* xc = xb + ((size_t)c << 12);
        const float* wcp = wj + c * 9;
#pragma unroll
        for (int dy = 0; dy < 3; ++dy) {
            int yy = h - 1 + dy;
            if ((unsigned)yy >= HDIM) continue;
#pragma unroll
            for (int dx = 0; dx < 3; ++dx) {
                int xx = wc - 1 + dx;
                if ((unsigned)xx >= WDIM) continue;
                acc = fmaf(xc[yy * WDIM + xx], wcp[dy * 3 + dx], acc);
            }
        }
    }
    om[((size_t)(b * 27 + j) << 12) + hw] = acc;
}

// ---------------------------------------------------------------------------
// Kernel 2: masked bilinear sampling -> im2col matrix A[m][c*9+kk] (f16).
// Threads contiguous in m (w inner) -> coalesced om reads, near-contiguous
// x corner gathers (x is L2-resident: 16.8 MB << 192 MB).
// ---------------------------------------------------------------------------
__global__ void sample_kernel(const float* __restrict__ x, const float* __restrict__ om,
                              _Float16* __restrict__ A)
{
    int m = blockIdx.x * 256 + threadIdx.x;    // gridDim.x = 64 -> 16384
    int c = blockIdx.y;                        // 0..255
    int b  = m >> 12;
    int hw = m & 4095;
    int h  = hw >> 6, wc = hw & 63;

    const float* xc  = x + ((size_t)(b * C_IN + c) << 12);
    const float* omb = om + ((size_t)(b * 27) << 12);
    _Float16* Arow = A + (size_t)m * KRED + c * KKPTS;

#pragma unroll
    for (int kk = 0; kk < KKPTS; ++kk) {
        float offy = omb[((size_t)kk << 12) + hw];
        float offx = omb[((size_t)(kk + KKPTS) << 12) + hw];
        float z    = omb[((size_t)(kk + 2 * KKPTS) << 12) + hw];
        float mask = 1.0f / (1.0f + __expf(-z));

        float py = (float)h - 1.0f + (float)(kk / 3) + offy;
        float px = (float)wc - 1.0f + (float)(kk % 3) + offx;
        float y0f = floorf(py), x0f = floorf(px);
        float ly = py - y0f, lx = px - x0f;
        int y0 = (int)y0f, x0 = (int)x0f;

        float v = 0.0f;
        float w00 = (1.0f - ly) * (1.0f - lx);
        float w01 = (1.0f - ly) * lx;
        float w10 = ly * (1.0f - lx);
        float w11 = ly * lx;
        if ((unsigned)y0 < HDIM && (unsigned)x0 < WDIM)
            v += xc[y0 * WDIM + x0] * w00;
        if ((unsigned)y0 < HDIM && (unsigned)(x0 + 1) < WDIM)
            v += xc[y0 * WDIM + x0 + 1] * w01;
        if ((unsigned)(y0 + 1) < HDIM && (unsigned)x0 < WDIM)
            v += xc[(y0 + 1) * WDIM + x0] * w10;
        if ((unsigned)(y0 + 1) < HDIM && (unsigned)(x0 + 1) < WDIM)
            v += xc[(y0 + 1) * WDIM + x0 + 1] * w11;

        Arow[kk] = (_Float16)(v * mask);
    }
}

// ---------------------------------------------------------------------------
// Kernel 3: WMMA GEMM  out[o][m] = sum_k Wf16[o][k] * A[m][k], fused BN+ReLU.
// - Register blocking: each wave keeps OTIL=4 accumulators (64 o rows per
//   block); every patch B-fragment is reused by 4 WMMAs.
// - Double-buffered LDS weight slab (2 x 32 KB): the next slab is fetched
//   into registers while WMMAs run on the current slab; one barrier/slab.
// 16-bit fragment striping per CDNA5 ISA: lane<16 -> K base 0, lane>=16 ->
// K base 8; v16h halves [0..7]=K kb..kb+7, [8..15]=K kb+16..kb+23.
// ---------------------------------------------------------------------------
__global__ __launch_bounds__(256) void gemm_kernel(const _Float16* __restrict__ Wf,
                                                   const _Float16* __restrict__ Ap,
                                                   const float* __restrict__ scale,
                                                   const float* __restrict__ shift,
                                                   float* __restrict__ out)
{
    __shared__ __align__(16) _Float16 Ws[2][16 * OTIL * KSLAB];   // 2 x 32 KB

    const int tid  = threadIdx.x;
    const int lane = tid & 31;
    const int wave = tid >> 5;
    const int o_base = blockIdx.y * (16 * OTIL);           // gridDim.y = 4
    const int m_tile = (blockIdx.x * 8 + wave) * 16;       // gridDim.x = 128

    const int row16 = lane & 15;
    const int kbase = (lane < 16) ? 0 : 8;

    const _Float16* wrow = Wf + (size_t)o_base * KRED;     // 64 contiguous rows
    const _Float16* prow = Ap + (size_t)(m_tile + row16) * KRED + kbase;

    // cooperative staging geometry: 2048 uint4/slab, 8 per thread,
    // q = tid + i*256 ; row = q>>5 ; col = q&31  (32 uint4 per 256-half row)
    const int q0 = tid;

    // preload slab 0 straight into LDS buffer 0
#pragma unroll
    for (int i = 0; i < 8; ++i) {
        int q = q0 + i * 256, r = q >> 5, ci = q & 31;
        ((uint4*)Ws[0])[q] = *((const uint4*)(wrow + (size_t)r * KRED) + ci);
    }
    __syncthreads();

    v8f acc[OTIL];
#pragma unroll
    for (int t = 0; t < OTIL; ++t) acc[t] = (v8f){};

    uint4 stg[8];

    for (int s = 0; s < NSLAB; ++s) {
        const int k0 = s * KSLAB;
        const _Float16* cur = Ws[s & 1];

        // fetch next slab into registers (latency hidden behind 32 WMMAs)
        if (s + 1 < NSLAB) {
#pragma unroll
            for (int i = 0; i < 8; ++i) {
                int q = q0 + i * 256, r = q >> 5, ci = q & 31;
                stg[i] = *((const uint4*)(wrow + (size_t)r * KRED + k0 + KSLAB) + ci);
            }
            __builtin_prefetch(prow + k0 + KSLAB, 0, 0);   // global_prefetch_b8
        }

#pragma unroll
        for (int kk = 0; kk < KSLAB; kk += 32) {
            // B fragment (patches) from global: loaded once, used by OTIL wmmas
            const _Float16* psrc = prow + k0 + kk;
            v8h blo = *(const v8h*)(psrc);
            v8h bhi = *(const v8h*)(psrc + 16);
            v16h bm;
#pragma unroll
            for (int i = 0; i < 8; ++i) { bm[i] = blo[i]; bm[i + 8] = bhi[i]; }

            // batch all OTIL A-fragment ds_loads before the WMMA chain
            v16h afrag[OTIL];
#pragma unroll
            for (int t = 0; t < OTIL; ++t) {
                const _Float16* wsrc = cur + (t * 16 + row16) * KSLAB + kk + kbase;
                v8h alo = *(const v8h*)(wsrc);
                v8h ahi = *(const v8h*)(wsrc + 16);
#pragma unroll
                for (int i = 0; i < 8; ++i) { afrag[t][i] = alo[i]; afrag[t][i + 8] = ahi[i]; }
            }
#pragma unroll
            for (int t = 0; t < OTIL; ++t)
                acc[t] = __builtin_amdgcn_wmma_f32_16x16x32_f16(
                    false, afrag[t], false, bm, (short)0, acc[t], false, false);
        }

        // dump staged slab into the other buffer; single barrier per slab
        if (s + 1 < NSLAB) {
#pragma unroll
            for (int i = 0; i < 8; ++i)
                ((uint4*)Ws[(s + 1) & 1])[q0 + i * 256] = stg[i];
            __syncthreads();
        }
    }

    // Epilogue: D lane col = m (contiguous w -> coalesced), row = o.
    const int mcol = m_tile + row16;
    const int bimg = mcol >> 12;
    const int hw   = mcol & 4095;
    const int ohalf = (lane < 16) ? 0 : 8;
#pragma unroll
    for (int t = 0; t < OTIL; ++t) {
#pragma unroll
        for (int r = 0; r < 8; ++r) {
            int o = o_base + t * 16 + ohalf + r;
            float v = acc[t][r] * scale[o] + shift[o];
            v = v > 0.0f ? v : 0.0f;
            out[(((size_t)bimg * O_OUT + o) << 12) + hw] = v;
        }
    }
}

// ---------------------------------------------------------------------------
extern "C" void kernel_launch(void* const* d_in, const int* in_sizes, int n_in,
                              void* d_out, int out_size, void* d_ws, size_t ws_size,
                              hipStream_t stream)
{
    const float* x     = (const float*)d_in[0];
    const float* w_off = (const float*)d_in[1];
    const float* b_off = (const float*)d_in[2];
    const float* w     = (const float*)d_in[3];
    const float* b     = (const float*)d_in[4];
    const float* gamma = (const float*)d_in[5];
    const float* beta  = (const float*)d_in[6];
    const float* rmean = (const float*)d_in[7];
    const float* rvar  = (const float*)d_in[8];
    float* out = (float*)d_out;

    // workspace carve-up (all 16B aligned)
    char* ws = (char*)d_ws;
    const size_t A_BYTES  = (size_t)MTOT * KRED * sizeof(_Float16);          // 75,497,472
    const size_t OM_BYTES = (size_t)BDIM * 27 * HDIM * WDIM * sizeof(float); // 1,769,472
    const size_t BT_BYTES = (size_t)O_OUT * KRED * sizeof(_Float16);         // 1,179,648
    _Float16* A  = (_Float16*)(ws);
    float*    om = (float*)(ws + A_BYTES);
    _Float16* BT = (_Float16*)(ws + A_BYTES + OM_BYTES);
    float* scale = (float*)(ws + A_BYTES + OM_BYTES + BT_BYTES);
    float* shift = scale + O_OUT;

    prep_kernel<<<(O_OUT * KRED + 255) / 256, 256, 0, stream>>>(
        w, b, gamma, beta, rmean, rvar, BT, scale, shift);

    offconv_kernel<<<dim3(16, 27, BDIM), 256, 0, stream>>>(x, w_off, b_off, om);

    sample_kernel<<<dim3(MTOT / 256, C_IN), 256, 0, stream>>>(x, om, A);

    gemm_kernel<<<dim3(MTOT / (8 * 16), O_OUT / (16 * OTIL)), 256, 0, stream>>>(
        BT, A, scale, shift, out);
}
